// ChebyshevKANLayer_30537217475172
// MI455X (gfx1250) — compile-verified, matrix operand order checked by reading, not run on hardware
//
#include <hip/hip_runtime.h>

// ChebyshevKAN layer as a WMMA GEMM on gfx1250 (wave32).
// y[b,o] = sum_i sum_j T_j(xn[b,i]) * C[i,o,j]
//   => GEMM: M=8192, N=1024, K=1024*9=9216, f16 operands, f32 accumulate.
//
// Pipeline:
//   K1: per-row min/max -> (scale, bias) in ws
//   K2: coeffs f32 [i][o][j] -> f16 K-major Bh[o][i*9+j] in ws (one-time,
//       L2-resident: 18.9 MB << 192 MB L2)
//   K3: 128x128 block tile, 8 waves as 4(M)x2(N), wave tile 32x64
//       (2x4 fragments of v_wmma_f32_16x16x32_f16), A tile = Chebyshev
//       recurrence computed on the fly, software-pipelined global loads.
//   Fallback (small ws): round-1 fused kernel (128x64 tile).

#define DEG      9
#define IN_DIM   1024
#define OUT_DIM  1024
#define NROWS    8192

#define KI  32           // i-values per K chunk
#define KC  (KI * DEG)   // 288 f16 K elements per chunk = 9 WMMA k-steps
#define SA  (KC + 8)     // padded LDS stride in halves (16B-aligned rows)

#define FBM 128          // fast-path block tile M
#define FBN 128          // fast-path block tile N
#define SBM 128          // fallback block tile M
#define SBN 64           // fallback block tile N

#define CI  32           // convert kernel: i per block
#define CO  64           // convert kernel: o per block

typedef __attribute__((ext_vector_type(16))) _Float16 v16h;
typedef __attribute__((ext_vector_type(8)))  float    v8f;
typedef __attribute__((ext_vector_type(4)))  float    v4f;
typedef __attribute__((ext_vector_type(4)))  unsigned int v4u;
typedef __attribute__((ext_vector_type(2)))  float    v2f;

// ---------------------------------------------------------------------------
// K1: per-row min/max -> affine (scale, bias) so xn = x*scale + bias
// ---------------------------------------------------------------------------
__global__ void __launch_bounds__(256)
cheby_row_stats(const float* __restrict__ x, float* __restrict__ stats) {
  __shared__ float smn[256];
  __shared__ float smx[256];
  const int row = blockIdx.x;
  const int tid = threadIdx.x;
  const float* xr = x + (size_t)row * IN_DIM;
  float mn = 3.4e38f, mx = -3.4e38f;
  for (int c = tid; c < IN_DIM; c += 256) {
    float v = xr[c];
    mn = fminf(mn, v);
    mx = fmaxf(mx, v);
  }
  smn[tid] = mn; smx[tid] = mx;
  __syncthreads();
  for (int s = 128; s > 0; s >>= 1) {
    if (tid < s) {
      smn[tid] = fminf(smn[tid], smn[tid + s]);
      smx[tid] = fmaxf(smx[tid], smx[tid + s]);
    }
    __syncthreads();
  }
  if (tid == 0) {
    float scale = 2.0f / (smx[0] - smn[0]);
    stats[2 * row + 0] = scale;
    stats[2 * row + 1] = -smn[0] * scale - 1.0f;
  }
}

// ---------------------------------------------------------------------------
// K2: one-time coeffs conversion + transpose to K-major f16:
//     Bh[o][i*9 + j] = (f16) C[i][o][j].
// Coalesced reads (576 consecutive floats per i), 576B-contiguous writes.
// ---------------------------------------------------------------------------
__global__ void __launch_bounds__(256)
cheby_convert_coeffs(const float* __restrict__ c, _Float16* __restrict__ bh) {
  __shared__ _Float16 t[CO][CI * DEG + 8];   // 64 x 296 halves
  const int i0 = blockIdx.x * CI;
  const int o0 = blockIdx.y * CO;
  const int tid = threadIdx.x;
  for (int idx = tid; idx < CI * CO * DEG; idx += 256) {
    int i  = idx / (CO * DEG);
    int t2 = idx % (CO * DEG);
    int o  = t2 / DEG;
    int j  = t2 % DEG;
    float v = c[(size_t)(i0 + i) * (OUT_DIM * DEG) + (size_t)(o0 + o) * DEG + j];
    t[o][i * DEG + j] = (_Float16)v;
  }
  __syncthreads();
  for (int idx = tid; idx < CO * (CI * DEG); idx += 256) {
    int o = idx / (CI * DEG);
    int k = idx % (CI * DEG);
    bh[(size_t)(o0 + o) * (IN_DIM * DEG) + (size_t)i0 * DEG + k] = t[o][k];
  }
}

// ---------------------------------------------------------------------------
// K3 (fast path): software-pipelined WMMA GEMM.
// ---------------------------------------------------------------------------
__global__ void __launch_bounds__(256, 1)
cheby_kan_gemm_fast(const float* __restrict__ x,
                    const _Float16* __restrict__ bh,
                    const float* __restrict__ stats,
                    float* __restrict__ out) {
  extern __shared__ _Float16 smem[];
  _Float16* As = smem;               // [FBM][SA]
  _Float16* Bs = smem + FBM * SA;    // [FBN][SA] (K-major rows, padded)

  const int tid  = threadIdx.x;
  const int lane = tid & 31;
  const int wave = tid >> 5;
  const int wm   = wave & 3;         // 0..3 : 32-row strip
  const int wn   = wave >> 2;        // 0..1 : 64-col strip
  const int m0   = blockIdx.x * FBM;
  const int n0   = blockIdx.y * FBN;
  const int lrow = lane & 15;
  const int lhi  = lane >> 4;

  v8f acc[2][4] = {};

  // Staging registers for the software pipeline.
  float xa[16];                      // x values for 16 (row, ii) pairs
  v4u   breg[18];                    // 18 x 16B of f16 B panel

  const int a_ii = tid & (KI - 1);   // this thread's i within chunk (fixed)
  const int a_r  = tid >> 5;         // base row (fixed); rows a_r + 8p

  auto load_chunk = [&](int i0c) {
#pragma unroll
    for (int p = 0; p < 16; ++p) {
      int r = a_r + p * 8;
      xa[p] = x[(size_t)(m0 + r) * IN_DIM + i0c + a_ii];
    }
#pragma unroll
    for (int p = 0; p < 18; ++p) {
      int idx = tid + p * 256;
      int o = idx / 36;              // B row (output channel), 0..127
      int u = idx % 36;              // 16B unit within the 576B row segment
      const v4u* src = (const v4u*)(bh + (size_t)(n0 + o) * (IN_DIM * DEG) +
                                    (size_t)i0c * DEG);
      breg[p] = src[u];
    }
  };

  auto store_chunk = [&]() {
#pragma unroll
    for (int p = 0; p < 16; ++p) {
      int r = a_r + p * 8;
      v2f sb = ((const v2f*)stats)[m0 + r];
      float xn = xa[p] * sb.x + sb.y;
      _Float16* dst = As + r * SA + a_ii * DEG;
      float t0 = 1.0f, t1 = xn;
      dst[0] = (_Float16)t0;
      dst[1] = (_Float16)t1;
#pragma unroll
      for (int n = 2; n < DEG; ++n) {
        float tn = 2.0f * xn * t1 - t0;
        dst[n] = (_Float16)tn;
        t0 = t1; t1 = tn;
      }
    }
#pragma unroll
    for (int p = 0; p < 18; ++p) {
      int idx = tid + p * 256;
      int o = idx / 36;
      int u = idx % 36;
      *(v4u*)(Bs + o * SA + u * 8) = breg[p];
    }
  };

  auto compute = [&]() {
#pragma unroll
    for (int ks = 0; ks < DEG; ++ks) {
      const int kofs = ks * 32 + lhi * 8;
      v16h afr[2], bfr[4];
#pragma unroll
      for (int fm = 0; fm < 2; ++fm) {
        const _Float16* p = As + (wm * 32 + fm * 16 + lrow) * SA + kofs;
        v4f* d = (v4f*)&afr[fm];
        d[0] = *(const v4f*)(p);
        d[1] = *(const v4f*)(p + 16);
      }
#pragma unroll
      for (int fn = 0; fn < 4; ++fn) {
        const _Float16* p = Bs + (wn * 64 + fn * 16 + lrow) * SA + kofs;
        v4f* d = (v4f*)&bfr[fn];
        d[0] = *(const v4f*)(p);
        d[1] = *(const v4f*)(p + 16);
      }
#pragma unroll
      for (int fm = 0; fm < 2; ++fm)
#pragma unroll
        for (int fn = 0; fn < 4; ++fn)
          acc[fm][fn] = __builtin_amdgcn_wmma_f32_16x16x32_f16(
              false, afr[fm], false, bfr[fn], (short)0, acc[fm][fn],
              false, false);
    }
  };

  const int nchunks = IN_DIM / KI;   // 32
  load_chunk(0);
  for (int c = 0; c < nchunks; ++c) {
    store_chunk();                    // regs (chunk c) -> LDS
    __syncthreads();
    if (c + 1 < nchunks) load_chunk((c + 1) * KI);  // overlap with WMMAs
    compute();                        // 9 k-steps x 8 WMMA
    __syncthreads();
  }

  // Epilogue: C/D layout -> VGPR v is row (v + 8*lhi), col = lane&15.
  const int mbase = m0 + wm * 32 + lhi * 8;
  const int nbase = n0 + wn * 64 + lrow;
#pragma unroll
  for (int fm = 0; fm < 2; ++fm)
#pragma unroll
    for (int fn = 0; fn < 4; ++fn)
#pragma unroll
      for (int v = 0; v < 8; ++v)
        out[(size_t)(mbase + fm * 16 + v) * OUT_DIM + (nbase + fn * 16)] =
            acc[fm][fn][v];
}

// ---------------------------------------------------------------------------
// Fallback: fused conversion GEMM (used if ws too small for Bh).
// ---------------------------------------------------------------------------
__global__ void __launch_bounds__(256)
cheby_kan_gemm_fused(const float* __restrict__ x,
                     const float* __restrict__ coeffs,
                     const float* __restrict__ stats,
                     float* __restrict__ out) {
  extern __shared__ _Float16 smem[];
  _Float16* As = smem;               // [SBM][SA]
  _Float16* Bs = smem + SBM * SA;    // [SBN][SA]

  const int tid  = threadIdx.x;
  const int lane = tid & 31;
  const int wave = tid >> 5;
  const int wm   = wave & 3;
  const int wn   = wave >> 2;
  const int m0   = blockIdx.x * SBM;
  const int n0   = blockIdx.y * SBN;
  const int lrow = lane & 15;
  const int lhi  = lane >> 4;

  v8f acc[2][2] = {};

  for (int i0 = 0; i0 < IN_DIM; i0 += KI) {
#pragma unroll
    for (int p = 0; p < (SBM * KI) / 256; ++p) {
      int idx = tid + p * 256;
      int ii  = idx & (KI - 1);
      int r   = idx >> 5;
      int grow = m0 + r;
      float s  = stats[2 * grow + 0];
      float bb = stats[2 * grow + 1];
      float xv = x[(size_t)grow * IN_DIM + i0 + ii];
      float xn = xv * s + bb;
      _Float16* dst = As + r * SA + ii * DEG;
      float t0 = 1.0f, t1 = xn;
      dst[0] = (_Float16)t0;
      dst[1] = (_Float16)t1;
#pragma unroll
      for (int n = 2; n < DEG; ++n) {
        float tn = 2.0f * xn * t1 - t0;
        dst[n] = (_Float16)tn;
        t0 = t1; t1 = tn;
      }
    }
#pragma unroll
    for (int p = 0; p < (KI * SBN) / 256; ++p) {
      int idx = tid + p * 256;
      int o   = idx & (SBN - 1);
      int ii  = idx >> 6;
      const float* src =
          coeffs + (size_t)(i0 + ii) * (OUT_DIM * DEG) + (size_t)(n0 + o) * DEG;
      _Float16* dst = Bs + o * SA + ii * DEG;
#pragma unroll
      for (int j = 0; j < DEG; ++j) dst[j] = (_Float16)src[j];
    }
    __syncthreads();

#pragma unroll
    for (int ks = 0; ks < DEG; ++ks) {
      const int kofs = ks * 32 + lhi * 8;
      v16h afr[2], bfr[2];
#pragma unroll
      for (int fm = 0; fm < 2; ++fm) {
        const _Float16* p = As + (wm * 32 + fm * 16 + lrow) * SA + kofs;
        v4f* d = (v4f*)&afr[fm];
        d[0] = *(const v4f*)(p);
        d[1] = *(const v4f*)(p + 16);
      }
#pragma unroll
      for (int fn = 0; fn < 2; ++fn) {
        const _Float16* p = Bs + (wn * 32 + fn * 16 + lrow) * SA + kofs;
        v4f* d = (v4f*)&bfr[fn];
        d[0] = *(const v4f*)(p);
        d[1] = *(const v4f*)(p + 16);
      }
#pragma unroll
      for (int fm = 0; fm < 2; ++fm)
#pragma unroll
        for (int fn = 0; fn < 2; ++fn)
          acc[fm][fn] = __builtin_amdgcn_wmma_f32_16x16x32_f16(
              false, afr[fm], false, bfr[fn], (short)0, acc[fm][fn],
              false, false);
    }
    __syncthreads();
  }

  const int mbase = m0 + wm * 32 + lhi * 8;
  const int nbase = n0 + wn * 32 + lrow;
#pragma unroll
  for (int fm = 0; fm < 2; ++fm)
#pragma unroll
    for (int fn = 0; fn < 2; ++fn)
#pragma unroll
      for (int v = 0; v < 8; ++v)
        out[(size_t)(mbase + fm * 16 + v) * OUT_DIM + (nbase + fn * 16)] =
            acc[fm][fn][v];
}

// ---------------------------------------------------------------------------
extern "C" void kernel_launch(void* const* d_in, const int* in_sizes, int n_in,
                              void* d_out, int out_size, void* d_ws,
                              size_t ws_size, hipStream_t stream) {
  (void)in_sizes; (void)n_in; (void)out_size;
  const float* x      = (const float*)d_in[0];
  const float* coeffs = (const float*)d_in[1];
  float*       stats  = (float*)d_ws;                 // 8192*2 floats = 64 KB
  float*       out    = (float*)d_out;

  cheby_row_stats<<<NROWS, 256, 0, stream>>>(x, stats);

  const size_t stats_bytes = (size_t)NROWS * 2 * sizeof(float);
  const size_t bh_bytes    = (size_t)OUT_DIM * IN_DIM * DEG * sizeof(_Float16);

  if (ws_size >= stats_bytes + bh_bytes) {
    _Float16* bh = (_Float16*)((char*)d_ws + stats_bytes);
    dim3 cgrid(IN_DIM / CI, OUT_DIM / CO);            // 32 x 16
    cheby_convert_coeffs<<<cgrid, 256, 0, stream>>>(coeffs, bh);

    dim3 grid(NROWS / FBM, OUT_DIM / FBN);            // 64 x 8
    size_t shmem = (size_t)(FBM + FBN) * SA * sizeof(_Float16);  // ~148 KB
    cheby_kan_gemm_fast<<<grid, 256, shmem, stream>>>(x, bh, stats, out);
  } else {
    dim3 grid(NROWS / SBM, OUT_DIM / SBN);            // 64 x 16
    size_t shmem = (size_t)(SBM + SBN) * SA * sizeof(_Float16);  // ~111 KB
    cheby_kan_gemm_fused<<<grid, 256, shmem, stream>>>(x, coeffs, stats, out);
  }
}
